// MLPF_23295902613680
// MI455X (gfx1250) — matrix operand order repaired and checked by reading, not compiled, not run on hardware
//
#include <hip/hip_runtime.h>
#include <hip/hip_bf16.h>
#include <math.h>

typedef __attribute__((ext_vector_type(16))) __bf16 v16bf;
typedef __attribute__((ext_vector_type(8)))  float  v8f;
typedef __attribute__((ext_vector_type(4)))  unsigned int u32x4;
typedef __attribute__((ext_vector_type(8)))  int          i32x8;
typedef __attribute__((ext_vector_type(4)))  int          i32x4;

#define N_TOT  65536
#define NPG    4096
#define G_CNT  16
#define DIM_IN 34
#define S_DIM  4
#define P_DIM  22
#define WIDTH  126
#define KNN    8

// ------------------------------ helpers -------------------------------------
__device__ __forceinline__ unsigned short f2bf(float f) {
  unsigned u = __float_as_uint(f);
  u += 0x7FFFu + ((u >> 16) & 1u);           // round-to-nearest-even
  return (unsigned short)(u >> 16);
}
__device__ __forceinline__ float bf2f(unsigned short h) {
  return __uint_as_float(((unsigned)h) << 16);
}
__device__ __forceinline__ float elu_f(float x) {
  return x > 0.f ? x : (__expf(x) - 1.f);
}
union FragU { uint4 q[2]; v16bf v; };        // 32B: one WMMA A/B fragment per lane

// ---------------------------------------------------------------------------
// 1) s = emb @ Ws + bs   (34->4),  h = emb @ Wh + bh  (34->22).  One thread/node.
// ---------------------------------------------------------------------------
__global__ __launch_bounds__(256)
void sh_kernel(const float* __restrict__ emb,
               const float* __restrict__ Ws, const float* __restrict__ bs,
               const float* __restrict__ Wh, const float* __restrict__ bh,
               float* __restrict__ s_out, float* __restrict__ h_out) {
  const int node = blockIdx.x * 256 + threadIdx.x;
  float e[DIM_IN];
#pragma unroll
  for (int i = 0; i < DIM_IN; ++i) e[i] = emb[(size_t)node * DIM_IN + i];
#pragma unroll
  for (int d = 0; d < S_DIM; ++d) {
    float acc = bs[d];
#pragma unroll
    for (int i = 0; i < DIM_IN; ++i) acc = fmaf(e[i], Ws[i * S_DIM + d], acc);
    s_out[(size_t)node * S_DIM + d] = acc;
  }
  for (int p = 0; p < P_DIM; ++p) {
    float acc = bh[p];
#pragma unroll
    for (int i = 0; i < DIM_IN; ++i) acc = fmaf(e[i], Wh[i * P_DIM + p], acc);
    h_out[(size_t)node * P_DIM + p] = acc;
  }
}

// ---------------------------------------------------------------------------
// 2) kNN: stage whole graph's coords (4096 x float4 = 64KB) into LDS with the
//    Tensor Data Mover (one descriptor, TENSORcnt-tracked), then each thread
//    scans all 4096 candidates keeping a sorted top-8 (smallest d2).
//    D# per CDNA5 ISA 8.3/8.4: group0 = {count=1, lds_addr, global_addr, type=2},
//    group1 = {data_size=8B, tensor_dim0=8192, tensor_dim1=1, tile_dim0=8192,
//              tile_dim1=1, tensor_dim0_stride=8192}.  2-D tensor -> groups 2/3
//    and the trailing group are disabled (zero).
// ---------------------------------------------------------------------------
__global__ __launch_bounds__(256)
void knn_kernel(const float* __restrict__ s,
                int* __restrict__ idx_out, float* __restrict__ d2k_out) {
  __shared__ __align__(16) float4 sc[NPG];
  const int g = blockIdx.y;
  const float* sg = s + (size_t)g * NPG * S_DIM;

  if (threadIdx.x < 32) {                       // wave 0 issues the TDM transfer
    const unsigned long long ga = (unsigned long long)(const void*)sg;
    const unsigned lds_addr = (unsigned)(unsigned long long)(void*)&sc[0];

    u32x4 g0;
    g0[0] = 1u;                                 // count=1, user descriptor
    g0[1] = lds_addr;                           // LDS destination (bytes)
    g0[2] = (unsigned)ga;                       // global_addr[31:0]
    g0[3] = (unsigned)((ga >> 32) & 0x01FFFFFFull) | 0x80000000u;  // [56:32] | type=2

    i32x8 g1;
    g1[0] = (int)(3u << 16);                    // data_size = 8B units
    g1[1] = (int)(8192u << 16);                 // tensor_dim0[15:0] into bits 47:32..63:48
    g1[2] = (int)(1u << 16);                    // tensor_dim0 hi=0 | tensor_dim1 lo=1
    g1[3] = (int)(8192u << 16);                 // tensor_dim1 hi=0 | tile_dim0=8192
    g1[4] = 1;                                  // tile_dim1=1, tile_dim2=0
    g1[5] = 8192;                               // tensor_dim0_stride lo
    g1[6] = 0;                                  // stride hi / dim1_stride lo
    g1[7] = 0;

    i32x4 z4 = {0, 0, 0, 0};
    i32x8 z8 = {0, 0, 0, 0, 0, 0, 0, 0};
    __builtin_amdgcn_tensor_load_to_lds(g0, g1, z4, z4, z8, 0);
    __builtin_amdgcn_s_wait_tensorcnt(0);       // s_wait_tensorcnt 0
  }
  __syncthreads();                              // release the other 7 waves

  const int ql = blockIdx.x * 256 + threadIdx.x;    // local query index
  const float4 p = sc[ql];
  float bd[KNN]; int bi[KNN];
#pragma unroll
  for (int t = 0; t < KNN; ++t) { bd[t] = 3.4e38f; bi[t] = 0; }

  for (int j = 0; j < NPG; ++j) {
    const float4 c = sc[j];
    const float dx = p.x - c.x, dy = p.y - c.y, dz = p.z - c.z, dw = p.w - c.w;
    const float d = fmaf(dx, dx, fmaf(dy, dy, fmaf(dz, dz, dw * dw)));
    if (d < bd[KNN - 1]) {
      bd[KNN - 1] = d; bi[KNN - 1] = j;
#pragma unroll
      for (int t = KNN - 1; t > 0; --t) {
        if (bd[t] < bd[t - 1]) {
          float td = bd[t]; bd[t] = bd[t - 1]; bd[t - 1] = td;
          int   ti = bi[t]; bi[t] = bi[t - 1]; bi[t - 1] = ti;
        }
      }
    }
  }
  const size_t node = (size_t)g * NPG + ql;
#pragma unroll
  for (int t = 0; t < KNN; ++t) {
    idx_out[node * KNN + t] = g * NPG + bi[t];
    d2k_out[node * KNN + t] = bd[t];
  }
}

// ---------------------------------------------------------------------------
// 3) aggregate (mean/max of w*h over 8 neighbors) + conv output GEMM 78->34.
// ---------------------------------------------------------------------------
__global__ __launch_bounds__(128)
void aggout_kernel(const float* __restrict__ emb_in, const float* __restrict__ h,
                   const int* __restrict__ idx, const float* __restrict__ d2k,
                   const float* __restrict__ Wo, const float* __restrict__ bo,
                   float* __restrict__ emb_out) {
  const size_t node = (size_t)blockIdx.x * 128 + threadIdx.x;
  float cat[DIM_IN + 2 * P_DIM];
#pragma unroll
  for (int i = 0; i < DIM_IN; ++i) cat[i] = emb_in[node * DIM_IN + i];
  float mean[P_DIM], mx[P_DIM];
#pragma unroll
  for (int p = 0; p < P_DIM; ++p) { mean[p] = 0.f; mx[p] = -3.4e38f; }
#pragma unroll
  for (int k = 0; k < KNN; ++k) {
    const int nb = idx[node * KNN + k];
    const float w = __expf(-10.f * d2k[node * KNN + k]);
    const float* hr = h + (size_t)nb * P_DIM;
#pragma unroll
    for (int p = 0; p < P_DIM; ++p) {
      const float m = hr[p] * w;
      mean[p] += m; mx[p] = fmaxf(mx[p], m);
    }
  }
#pragma unroll
  for (int p = 0; p < P_DIM; ++p) {
    cat[DIM_IN + p]         = mean[p] * (1.f / KNN);
    cat[DIM_IN + P_DIM + p] = mx[p];
  }
  for (int o = 0; o < DIM_IN; ++o) {
    float acc = bo[o];
#pragma unroll
    for (int c = 0; c < DIM_IN + 2 * P_DIM; ++c)
      acc = fmaf(cat[c], Wo[c * DIM_IN + o], acc);
    emb_out[node * DIM_IN + o] = acc;
  }
}

// ---------------------------------------------------------------------------
// 4) Pre-swizzle head weights into bf16 WMMA B-fragment layout.
//    Per head: layer0 [64x128] (16 tiles), layer1/2 [128x128] (32 tiles each).
//    Tile(kt,nt) = 1024B block; lane l holds 32B = 16 bf16 of column nt*16+(l&15),
//    K = kt*32 + (l>=16 ? 16:0) + 0..15   (contiguous K order, same as the A loads).
// ---------------------------------------------------------------------------
__global__ __launch_bounds__(512)
void prep_weights(const float* W0a, const float* Wha,
                  const float* W0b, const float* Whb,
                  const float* W0c, const float* Whc,
                  unsigned short* __restrict__ wpack) {
  const int head = blockIdx.x / 80;
  const int t    = blockIdx.x % 80;
  const float* W0 = (head == 0) ? W0a : (head == 1) ? W0b : W0c;
  const float* Wh = (head == 0) ? Wha : (head == 1) ? Whb : Whc;

  int kt, nt, K; const float* src;
  if (t < 16)       { kt = t >> 3;        nt = t & 7; src = W0;                K = DIM_IN; }
  else if (t < 48)  { int tt = t - 16; kt = tt >> 3; nt = tt & 7; src = Wh;    K = WIDTH;  }
  else              { int tt = t - 48; kt = tt >> 3; nt = tt & 7; src = Wh + WIDTH * WIDTH; K = WIDTH; }

  const int e    = threadIdx.x;          // 0..511 : one bf16 element of the tile
  const int lane = e >> 4;
  const int kk   = e & 15;
  const int k    = kt * 32 + ((lane >= 16) ? 16 : 0) + kk;
  const int col  = nt * 16 + (lane & 15);
  const float v  = (k < K && col < WIDTH) ? src[k * WIDTH + col] : 0.f;
  wpack[(size_t)head * 40960 + (size_t)t * 512 + lane * 16 + kk] = f2bf(v);
}

// ---------------------------------------------------------------------------
// 5) Fused MLP head via v_wmma_f32_16x16x32_bf16.
//    Block = 128 thr (4 waves); each wave owns a 16-row tile -> 64 rows/block.
//    Activations ping-pong in LDS (bf16, stride 136 = 272B = 17*16B aligned).
// ---------------------------------------------------------------------------
#define ACT_STRIDE 136
__global__ __launch_bounds__(128)
void head_kernel(const float* __restrict__ emb,
                 const unsigned short* __restrict__ wpack,   // this head's packed weights
                 const float* __restrict__ b0, const float* __restrict__ bh,
                 const float* __restrict__ Wf, const float* __restrict__ bf,
                 const float* __restrict__ x, float* __restrict__ out,
                 int out_dim, int add_x) {
  __shared__ __align__(16) unsigned short acts[2][64 * ACT_STRIDE];
  const int tid  = threadIdx.x;
  const int wave = tid >> 5;
  const int lane = tid & 31;
  const int row0 = blockIdx.x * 64;

  // stage input rows -> bf16 acts[0], pad K 34..63 with zeros
  for (int e = tid; e < 64 * 64; e += 128) {
    const int r = e >> 6, c = e & 63;
    const float v = (c < DIM_IN) ? emb[(size_t)(row0 + r) * DIM_IN + c] : 0.f;
    acts[0][r * ACT_STRIDE + c] = f2bf(v);
  }
  __syncthreads();

  int cur = 0;
  const unsigned short* wlayer = wpack;
  const int arow  = lane & 15;
  const int khalf = lane >> 4;       // A-frag: lane<16 -> K 0-15 ; lane>=16 -> 16-31
  const int ocol_lo  = lane & 15;
  const int orow_off = (lane >= 16) ? 8 : 0;

  for (int l = 0; l < 3; ++l) {
    const int nkt = (l == 0) ? 2 : 4;
    const float* bias = (l == 0) ? b0 : (bh + (l - 1) * WIDTH);
    const unsigned short* abase = &acts[cur][(wave * 16 + arow) * ACT_STRIDE];

    // hoist A fragments for all K tiles of this wave's 16-row slab
    FragU afr[4];
#pragma unroll
    for (int kt = 0; kt < 4; ++kt)
      if (kt < nkt) {
        afr[kt].q[0] = *(const uint4*)(abase + kt * 32 + khalf * 16);
        afr[kt].q[1] = *(const uint4*)(abase + kt * 32 + khalf * 16 + 8);
      }

    for (int nt = 0; nt < 8; ++nt) {
      const int c = nt * 16 + ocol_lo;
      const float bv = (c < WIDTH) ? bias[c] : 0.f;
      v8f acc;
#pragma unroll
      for (int r = 0; r < 8; ++r) acc[r] = bv;
#pragma unroll
      for (int kt = 0; kt < 4; ++kt)
        if (kt < nkt) {
          FragU bfr;
          const unsigned short* bp = wlayer + ((size_t)(kt * 8 + nt)) * 512 + lane * 16;
          bfr.q[0] = *(const uint4*)(bp);
          bfr.q[1] = *(const uint4*)(bp + 8);
          acc = __builtin_amdgcn_wmma_f32_16x16x32_bf16(
              /*neg_a=*/false, afr[kt].v, /*neg_b=*/false, bfr.v,
              /*c_mod=*/(short)0, acc, /*reuse_a=*/false, /*reuse_b=*/false);
        }
      // ELU + store bf16 activation for next layer
#pragma unroll
      for (int r = 0; r < 8; ++r) {
        const int orow = wave * 16 + orow_off + r;
        acts[1 - cur][orow * ACT_STRIDE + c] = f2bf(elu_f(acc[r]));
      }
    }
    __syncthreads();
    cur ^= 1;
    wlayer += (size_t)nkt * 8 * 512;
  }

  // final thin layer 126 -> out_dim in fp32 (out_dim <= 8)
  for (int o = tid; o < 64 * out_dim; o += 128) {
    const int r = o / out_dim, j = o % out_dim;
    float acc = bf[j];
    const unsigned short* hrow = &acts[cur][r * ACT_STRIDE];
#pragma unroll 7
    for (int c = 0; c < WIDTH; ++c)
      acc = fmaf(bf2f(hrow[c]), Wf[c * out_dim + j], acc);
    if (add_x) acc += x[(size_t)(row0 + r) * DIM_IN + 1 + j];
    out[(size_t)(row0 + r) * out_dim + j] = acc;
  }
}

// ---------------------------------------------------------------------------
extern "C" void kernel_launch(void* const* d_in, const int* in_sizes, int n_in,
                              void* d_out, int out_size, void* d_ws, size_t ws_size,
                              hipStream_t stream) {
  const float* x        = (const float*)d_in[0];
  const float* conv_Ws  = (const float*)d_in[1];
  const float* conv_bs  = (const float*)d_in[2];
  const float* conv_Wh  = (const float*)d_in[3];
  const float* conv_bh  = (const float*)d_in[4];
  const float* conv_Wo  = (const float*)d_in[5];
  const float* conv_bo  = (const float*)d_in[6];
  const float* hW0[3] = { (const float*)d_in[7],  (const float*)d_in[13], (const float*)d_in[19] };
  const float* hb0[3] = { (const float*)d_in[8],  (const float*)d_in[14], (const float*)d_in[20] };
  const float* hWh[3] = { (const float*)d_in[9],  (const float*)d_in[15], (const float*)d_in[21] };
  const float* hbh[3] = { (const float*)d_in[10], (const float*)d_in[16], (const float*)d_in[22] };
  const float* hWf[3] = { (const float*)d_in[11], (const float*)d_in[17], (const float*)d_in[23] };
  const float* hbf[3] = { (const float*)d_in[12], (const float*)d_in[18], (const float*)d_in[24] };

  // ---- workspace carve-up (bytes) ----
  char* ws = (char*)d_ws;
  unsigned short* wpack = (unsigned short*)ws;               //   245,760 B
  float* s_buf   = (float*)(ws +   262144);                  // N*4*4   = 1,048,576
  float* h_buf   = (float*)(ws +  1310720);                  // N*22*4  = 5,767,168
  int*   idx_buf = (int*)  (ws +  7077888);                  // N*8*4   = 2,097,152
  float* d2k_buf = (float*)(ws +  9175040);                  // N*8*4   = 2,097,152
  float* embA    = (float*)(ws + 11272192);                  // N*34*4  = 8,912,896
  float* embB    = (float*)(ws + 20185088);                  // N*34*4  = 8,912,896

  // pack head weights into bf16 WMMA fragment layout
  prep_weights<<<240, 512, 0, stream>>>(hW0[0], hWh[0], hW0[1], hWh[1], hW0[2], hWh[2], wpack);

  // two GravNet convolutions
  const float* emb_in = x;
  float* emb_out = embA;
  for (int i = 0; i < 2; ++i) {
    sh_kernel<<<N_TOT / 256, 256, 0, stream>>>(
        emb_in,
        conv_Ws + (size_t)i * DIM_IN * S_DIM, conv_bs + (size_t)i * S_DIM,
        conv_Wh + (size_t)i * DIM_IN * P_DIM, conv_bh + (size_t)i * P_DIM,
        s_buf, h_buf);
    knn_kernel<<<dim3(NPG / 256, G_CNT), 256, 0, stream>>>(s_buf, idx_buf, d2k_buf);
    aggout_kernel<<<N_TOT / 128, 128, 0, stream>>>(
        emb_in, h_buf, idx_buf, d2k_buf,
        conv_Wo + (size_t)i * (DIM_IN + 2 * P_DIM) * DIM_IN,
        conv_bo + (size_t)i * DIM_IN, emb_out);
    emb_in = emb_out;
    emb_out = embB;
  }

  // three heads: id (8 cls), reg (4, +x[:,1:5]), chg (1)
  float* out_f = (float*)d_out;
  const int   odim[3] = { 8, 4, 1 };
  const int   addx[3] = { 0, 1, 0 };
  size_t ooff = 0;
  for (int hId = 0; hId < 3; ++hId) {
    head_kernel<<<N_TOT / 64, 128, 0, stream>>>(
        emb_in, wpack + (size_t)hId * 40960,
        hb0[hId], hbh[hId], hWf[hId], hbf[hId],
        x, out_f + ooff, odim[hId], addx[hId]);
    ooff += (size_t)N_TOT * odim[hId];
  }
}
